// LVM_87952340287896
// MI455X (gfx1250) — matrix operand order repaired and checked by reading, not compiled, not run on hardware
//
#include <hip/hip_runtime.h>
#include <hip/hip_bf16.h>

// ---------------- problem constants ----------------
#define Bz   8
#define Sz   16
#define HWz  16384          // H*W = 128*128
#define Ez   640
#define E3z  1920           // 3*E
#define NHz  10
#define HDz  64
#define Lz   22
#define FFz  2048
#define SCALEF 0.125f       // 1/sqrt(64)

// ---------------- WMMA fragment types (ISA 7.12.2) ----------------
typedef __bf16 bf16_t;
typedef bf16_t v16bf __attribute__((ext_vector_type(16)));
typedef float  v8f   __attribute__((ext_vector_type(8)));

union ABFrag { v16bf v; uint4 q[2]; };
union CFrag  { v8f  v; float f[8]; };

__device__ __forceinline__ unsigned short f32_to_bf16(float f) {
    unsigned int x = __float_as_uint(f);
    unsigned int r = x + 0x7FFFu + ((x >> 16) & 1u);   // round-to-nearest-even
    return (unsigned short)(r >> 16);
}
__device__ __forceinline__ unsigned int pack_bf16x2(float lo, float hi) {
    return (unsigned int)f32_to_bf16(lo) | ((unsigned int)f32_to_bf16(hi) << 16);
}

// A-fragment pack layout, shared by all producers/consumers:
//   element (m, k), k even pair (k, k+1), K multiple of 32
//   kstep = k>>5 ; off = k&31 ; half = (off>>3)&1 ; v = ((off>>4)<<2) + ((off&7)>>1)
//   uint index = kstep*256 + (half*16 + m)*8 + v
__device__ __forceinline__ int pk_idx(int m, int k) {
    const int kstep = k >> 5, off = k & 31;
    const int half = (off >> 3) & 1;
    const int v    = ((off >> 4) << 2) + ((off & 7) >> 1);
    return kstep * 256 + (half * 16 + m) * 8 + v;
}

// ---------------------------------------------------------------------------
// pack_a: A_fp32[16, K] (row remap m -> (m*aMul+aAdd)*strideA, rows >= Mvalid
// zero-filled) -> WMMA A-fragment-ordered bf16 pairs. Mvalid=0 => zero fill.
// ---------------------------------------------------------------------------
__global__ void pack_a_kernel(const float* __restrict__ A, int strideA,
                              int aMul, int aAdd, int Mvalid, int K,
                              unsigned int* __restrict__ out)
{
    const int total = (K >> 5) * 256;
    for (int i = blockIdx.x * blockDim.x + threadIdx.x; i < total;
         i += gridDim.x * blockDim.x) {
        const int kstep = i >> 8;
        const int lane  = (i >> 3) & 31;
        const int v     = i & 7;
        const int m     = lane & 15;
        const int half  = lane >> 4;
        const int k     = kstep * 32 + half * 8 + ((v < 4) ? 2 * v : 16 + 2 * (v - 4));
        unsigned int u = 0u;
        if (m < Mvalid) {
            const float2 f = *(const float2*)(A + (size_t)(m * aMul + aAdd) * (size_t)strideA + k);
            u = pack_bf16x2(f.x, f.y);
        }
        out[i] = u;
    }
}

// ---------------------------------------------------------------------------
// Skinny GEMM: D[16, N] = Apack[16,K](bf16 frag order) @ W_bf16[N,K]^T
// grid.x = N/16 tiles, grid.y = split-K; block = 128 = 4 waves (K quarters),
// LDS-reduced, wave 0 epilogue. Inner loop: 4x b128 loads + 1 WMMA.
// mode 0: store fp32 (bias/bias2/resid/relu) and/or packed bf16 out
// mode 1: atomicAdd fp32 (split-K)
// ---------------------------------------------------------------------------
__global__ __launch_bounds__(128) void gemm16_wmma(
    const unsigned int* __restrict__ Apack,
    const unsigned short* __restrict__ W, int K, int kChunk,
    const float* __restrict__ bias, const float* __restrict__ bias2,
    const float* __restrict__ resid, int strideR,
    float* __restrict__ Out, int strideO, int oMul, int oAdd, int Mvalid,
    unsigned int* __restrict__ packOut, int mode, int relu)
{
    const int lane   = threadIdx.x & 31;
    const int wave   = threadIdx.x >> 5;
    const int n0     = blockIdx.x * 16;
    const int kwave  = kChunk >> 2;
    const int kBase  = blockIdx.y * kChunk + wave * kwave;
    const int halfHi = lane >> 4;
    const int nl     = lane & 15;

    const uint4* Ap = (const uint4*)Apack + (size_t)(kBase >> 5) * 64 + lane * 2;
    const uint4* Bp = (const uint4*)(W + (size_t)(n0 + nl) * (size_t)K + kBase + halfHi * 8);

    CFrag c;
    #pragma unroll
    for (int i = 0; i < 8; ++i) c.f[i] = 0.0f;

    const int steps = kwave >> 5;
    #pragma unroll 2
    for (int s = 0; s < steps; ++s) {
        ABFrag a, b;
        a.q[0] = Ap[0];
        a.q[1] = Ap[1];
        b.q[0] = Bp[0];
        b.q[1] = Bp[2];
        c.v = __builtin_amdgcn_wmma_f32_16x16x32_bf16(
                  false, a.v, false, b.v, (short)0, c.v, false, false);
        Ap += 64;   // 256 uints per k-step
        Bp += 4;    // 32 bf16 per k-step
    }

    __shared__ float red[4][8][32];
    #pragma unroll
    for (int r = 0; r < 8; ++r) red[wave][r][lane] = c.f[r];
    __syncthreads();
    if (wave != 0) return;

    #pragma unroll
    for (int r = 0; r < 8; ++r) {
        float val = red[0][r][lane] + red[1][r][lane] + red[2][r][lane] + red[3][r][lane];
        const int mr = r + halfHi * 8;
        const int n  = n0 + nl;
        if (mode == 1) {
            if (mr < Mvalid)
                atomicAdd(&Out[(size_t)(mr * oMul + oAdd) * (size_t)strideO + n], val);
        } else {
            if (bias)  val += bias[n];
            if (bias2) val += bias2[n];
            if (resid) val += resid[(size_t)mr * (size_t)strideR + n];
            if (relu)  val = fmaxf(val, 0.0f);
            if (Out && mr < Mvalid)
                Out[(size_t)(mr * oMul + oAdd) * (size_t)strideO + n] = val;
            if (packOut) {
                // pair (n, n+1) lives in adjacent lanes of the same half
                const float vn = __shfl_xor(val, 1, 32);
                if (!(nl & 1) && mr < Mvalid)
                    packOut[pk_idx(mr, n)] = pack_bf16x2(val, vn);
            }
        }
    }
}

// ---------------------------------------------------------------------------
// Attention: 2-token, 10-head; one wave per (batch, head). Emits the context
// projection input directly as packed bf16 A-fragments.
// ---------------------------------------------------------------------------
__global__ __launch_bounds__(32) void attn2_kernel(const float* __restrict__ qkv,
                                                   unsigned int* __restrict__ opack)
{
    const int b = blockIdx.x / NHz, h = blockIdx.x % NHz;
    const int lane = threadIdx.x;
    const float* r0 = qkv + (size_t)(2 * b)     * E3z + h * HDz;
    const float* r1 = qkv + (size_t)(2 * b + 1) * E3z + h * HDz;

    const float q0a = r0[lane],        q0b = r0[lane + 32];
    const float q1a = r1[lane],        q1b = r1[lane + 32];
    const float k0a = r0[Ez + lane],   k0b = r0[Ez + lane + 32];
    const float k1a = r1[Ez + lane],   k1b = r1[Ez + lane + 32];
    const float v0a = r0[2*Ez + lane], v0b = r0[2*Ez + lane + 32];
    const float v1a = r1[2*Ez + lane], v1b = r1[2*Ez + lane + 32];

    float s00 = q0a*k0a + q0b*k0b, s01 = q0a*k1a + q0b*k1b;
    float s10 = q1a*k0a + q1b*k0b, s11 = q1a*k1a + q1b*k1b;
    #pragma unroll
    for (int off = 16; off > 0; off >>= 1) {
        s00 += __shfl_xor(s00, off, 32);
        s01 += __shfl_xor(s01, off, 32);
        s10 += __shfl_xor(s10, off, 32);
        s11 += __shfl_xor(s11, off, 32);
    }
    s00 *= SCALEF; s01 *= SCALEF; s10 *= SCALEF; s11 *= SCALEF;
    const float m0 = fmaxf(s00, s01), m1 = fmaxf(s10, s11);
    const float e00 = __expf(s00 - m0), e01 = __expf(s01 - m0);
    const float e10 = __expf(s10 - m1), e11 = __expf(s11 - m1);
    const float i0 = 1.0f / (e00 + e01), i1 = 1.0f / (e10 + e11);
    const float a00 = e00 * i0, a01 = e01 * i0;
    const float a10 = e10 * i1, a11 = e11 * i1;

    const float o0a = a00 * v0a + a01 * v1a, o0b = a00 * v0b + a01 * v1b;
    const float o1a = a10 * v0a + a11 * v1a, o1b = a10 * v0b + a11 * v1b;

    const float p0a = __shfl_xor(o0a, 1, 32), p0b = __shfl_xor(o0b, 1, 32);
    const float p1a = __shfl_xor(o1a, 1, 32), p1b = __shfl_xor(o1b, 1, 32);
    if (!(lane & 1)) {
        const int kA = h * HDz + lane, kB = kA + 32;
        opack[pk_idx(2 * b,     kA)] = pack_bf16x2(o0a, p0a);
        opack[pk_idx(2 * b,     kB)] = pack_bf16x2(o0b, p0b);
        opack[pk_idx(2 * b + 1, kA)] = pack_bf16x2(o1a, p1a);
        opack[pk_idx(2 * b + 1, kB)] = pack_bf16x2(o1b, p1b);
    }
}

// ---------------------------------------------------------------------------
// LayerNorm over E=640, one block per row; writes fp32 row + packed bf16 row.
// ---------------------------------------------------------------------------
__global__ __launch_bounds__(256) void ln_kernel(const float* __restrict__ in,
                                                 const float* __restrict__ w,
                                                 const float* __restrict__ b,
                                                 float* __restrict__ out,
                                                 unsigned int* __restrict__ pk)
{
    const int row = blockIdx.x;
    const float* x = in + (size_t)row * Ez;
    __shared__ float s1[256], s2[256];
    float sum = 0.0f, sq = 0.0f;
    for (int i = threadIdx.x; i < Ez; i += 256) { const float v = x[i]; sum += v; sq += v * v; }
    s1[threadIdx.x] = sum; s2[threadIdx.x] = sq;
    __syncthreads();
    for (int st = 128; st > 0; st >>= 1) {
        if (threadIdx.x < st) { s1[threadIdx.x] += s1[threadIdx.x + st];
                                s2[threadIdx.x] += s2[threadIdx.x + st]; }
        __syncthreads();
    }
    const float mean = s1[0] * (1.0f / Ez);
    const float var  = s2[0] * (1.0f / Ez) - mean * mean;
    const float inv  = rsqrtf(var + 1e-5f);
    for (int j = threadIdx.x; j < (Ez >> 1); j += 256) {
        const int k = 2 * j;
        const float v0 = (x[k]     - mean) * inv * w[k]     + b[k];
        const float v1 = (x[k + 1] - mean) * inv * w[k + 1] + b[k + 1];
        out[(size_t)row * Ez + k]     = v0;
        out[(size_t)row * Ez + k + 1] = v1;
        pk[pk_idx(row, k)] = pack_bf16x2(v0, v1);
    }
}

// ---------------- glue kernels ----------------
__global__ void cvt_bf16_kernel(const float* __restrict__ src,
                                unsigned short* __restrict__ dst, int n)
{
    for (int i = blockIdx.x * blockDim.x + threadIdx.x; i < n;
         i += gridDim.x * blockDim.x)
        dst[i] = f32_to_bf16(src[i]);
}

// x[2b, :] = slice_b  (init for split-K atomic slice GEMM)
__global__ void init_even_rows(float* __restrict__ x, const float* __restrict__ b) {
    int i = blockIdx.x * blockDim.x + threadIdx.x;
    if (i < Bz * Ez) { int r = i / Ez, e = i % Ez; x[(size_t)(2 * r) * Ez + e] = b[e]; }
}

// ---------------------------------------------------------------------------
extern "C" void kernel_launch(void* const* d_in, const int* in_sizes, int n_in,
                              void* d_out, int out_size, void* d_ws, size_t ws_size,
                              hipStream_t stream)
{
    (void)in_sizes; (void)n_in; (void)out_size; (void)ws_size;

    const float* slices   = (const float*)d_in[0];
    const float* pos      = (const float*)d_in[1];
    const float* slice_w  = (const float*)d_in[2];
    const float* slice_b  = (const float*)d_in[3];
    const float* hidden_w = (const float*)d_in[4];
    const float* hidden_b = (const float*)d_in[5];
    const float* in_w     = (const float*)d_in[6];
    const float* in_b     = (const float*)d_in[7];
    const float* aow      = (const float*)d_in[8];
    const float* aob      = (const float*)d_in[9];
    const float* ln1w     = (const float*)d_in[10];
    const float* ln1b     = (const float*)d_in[11];
    const float* f1w      = (const float*)d_in[12];
    const float* f1b      = (const float*)d_in[13];
    const float* f2w      = (const float*)d_in[14];
    const float* f2b      = (const float*)d_in[15];
    const float* ln2w     = (const float*)d_in[16];
    const float* ln2b     = (const float*)d_in[17];
    const float* out_w    = (const float*)d_in[18];
    const float* out_b    = (const float*)d_in[19];
    float* out = (float*)d_out;

    // ---- workspace carve-out ----
    size_t off = 0;
    auto carve = [&](size_t bytes) -> char* {
        char* p = (char*)d_ws + off;
        off += (bytes + 255) & ~(size_t)255;
        return p;
    };
    unsigned short* wb_slice = (unsigned short*)carve((size_t)Ez * HWz * 2);
    unsigned short* wb_hid   = (unsigned short*)carve((size_t)Ez * Ez * 2);
    unsigned short* wb_in    = (unsigned short*)carve((size_t)Lz * E3z * Ez * 2);
    unsigned short* wb_ao    = (unsigned short*)carve((size_t)Lz * Ez * Ez * 2);
    unsigned short* wb_f1    = (unsigned short*)carve((size_t)Lz * FFz * Ez * 2);
    unsigned short* wb_f2    = (unsigned short*)carve((size_t)Lz * Ez * FFz * 2);
    unsigned short* wb_out   = (unsigned short*)carve((size_t)HWz * Ez * 2);
    float* xbuf   = (float*)carve((size_t)16 * Ez * 4);
    float* qkvbuf = (float*)carve((size_t)16 * E3z * 4);
    float* r1buf  = (float*)carve((size_t)16 * Ez * 4);
    unsigned int* xpack   = (unsigned int*)carve((size_t)(Ez  >> 5) * 256 * 4);
    unsigned int* opack   = (unsigned int*)carve((size_t)(Ez  >> 5) * 256 * 4);
    unsigned int* hpack   = (unsigned int*)carve((size_t)(FFz >> 5) * 256 * 4);
    unsigned int* slpack  = (unsigned int*)carve((size_t)(HWz >> 5) * 256 * 4);
    unsigned int* prpack  = (unsigned int*)carve((size_t)(Ez  >> 5) * 256 * 4);
    unsigned int* hidpack = (unsigned int*)carve((size_t)(Ez  >> 5) * 256 * 4);

    // ---- one-time (per call) fp32 -> bf16 weight conversion ----
    auto cvt = [&](const float* s, unsigned short* d, size_t n) {
        cvt_bf16_kernel<<<4096, 256, 0, stream>>>(s, d, (int)n);
    };
    cvt(slice_w,  wb_slice, (size_t)Ez * HWz);
    cvt(hidden_w, wb_hid,   (size_t)Ez * Ez);
    cvt(in_w,     wb_in,    (size_t)Lz * E3z * Ez);
    cvt(aow,      wb_ao,    (size_t)Lz * Ez * Ez);
    cvt(f1w,      wb_f1,    (size_t)Lz * FFz * Ez);
    cvt(f2w,      wb_f2,    (size_t)Lz * Ez * FFz);
    cvt(out_w,    wb_out,   (size_t)HWz * Ez);

    auto pack = [&](const float* A, int strideA, int aMul, int aAdd, int Mv,
                    int K, unsigned int* dst) {
        const int total = (K >> 5) * 256;
        pack_a_kernel<<<(total + 255) / 256, 256, 0, stream>>>(
            A, strideA, aMul, aAdd, Mv, K, dst);
    };
    auto gemm = [&](const unsigned int* Ap, const unsigned short* W, int N, int K,
                    int kChunk, int ksplit,
                    const float* bias, const float* bias2,
                    const float* resid, int strideR,
                    float* Out, int strideO, int oMul, int oAdd, int Mv,
                    unsigned int* packOut, int mode, int relu) {
        gemm16_wmma<<<dim3(N / 16, ksplit), dim3(128), 0, stream>>>(
            Ap, W, K, kChunk, bias, bias2, resid, strideR,
            Out, strideO, oMul, oAdd, Mv, packOut, mode, relu);
    };

    // hidden state starts at zero -> zero-filled packed fragments
    pack(xbuf, Ez, 1, 0, /*Mvalid=*/0, Ez, hidpack);

    for (int s = 0; s < Sz; ++s) {
        // x[2b] = slice_b + slices[b,s] @ slice_w.T   (split-K=8, atomic)
        init_even_rows<<<(Bz * Ez + 255) / 256, 256, 0, stream>>>(xbuf, slice_b);
        pack(slices + (size_t)s * HWz, Sz * HWz, 1, 0, Bz, HWz, slpack);
        gemm(slpack, wb_slice, Ez, HWz, HWz / 8, 8,
             nullptr, nullptr, nullptr, 0,
             xbuf, Ez, 2, 0, Bz, nullptr, /*mode=*/1, 0);
        // x[2b+1] = hidden @ hidden_w.T + hidden_b + pos
        gemm(hidpack, wb_hid, Ez, Ez, Ez, 1,
             hidden_b, pos, nullptr, 0,
             xbuf, Ez, 2, 1, Bz, nullptr, 0, 0);
        // packed x for the first qkv GEMM
        pack(xbuf, Ez, 1, 0, 16, Ez, xpack);

        for (int l = 0; l < Lz; ++l) {
            // qkv = x @ in_w[l].T + in_b[l]
            gemm(xpack, wb_in + (size_t)l * E3z * Ez, E3z, Ez, Ez, 1,
                 in_b + (size_t)l * E3z, nullptr, nullptr, 0,
                 qkvbuf, E3z, 1, 0, 16, nullptr, 0, 0);
            // attention -> packed o
            attn2_kernel<<<Bz * NHz, 32, 0, stream>>>(qkvbuf, opack);
            // r1 = x + o @ aow[l].T + aob[l]
            gemm(opack, wb_ao + (size_t)l * Ez * Ez, Ez, Ez, Ez, 1,
                 aob + (size_t)l * Ez, nullptr, xbuf, Ez,
                 r1buf, Ez, 1, 0, 16, nullptr, 0, 0);
            // x = LN1(r1)  (+ packed x)
            ln_kernel<<<16, 256, 0, stream>>>(r1buf, ln1w + (size_t)l * Ez,
                                              ln1b + (size_t)l * Ez, xbuf, xpack);
            // hpack = pack(relu(x @ f1w[l].T + f1b[l]))   (no fp32 output)
            gemm(xpack, wb_f1 + (size_t)l * FFz * Ez, FFz, Ez, Ez, 1,
                 f1b + (size_t)l * FFz, nullptr, nullptr, 0,
                 nullptr, 0, 1, 0, 16, hpack, 0, /*relu=*/1);
            // r1 = x + h @ f2w[l].T + f2b[l]
            gemm(hpack, wb_f2 + (size_t)l * Ez * FFz, Ez, FFz, FFz, 1,
                 f2b + (size_t)l * Ez, nullptr, xbuf, Ez,
                 r1buf, Ez, 1, 0, 16, nullptr, 0, 0);
            // x = LN2(r1)  (+ packed x)
            ln_kernel<<<16, 256, 0, stream>>>(r1buf, ln2w + (size_t)l * Ez,
                                              ln2b + (size_t)l * Ez, xbuf, xpack);
        }

        // preds[b, s, :] = t[:,0] @ out_w.T + out_b   (A row m -> x row 2m)
        pack(xbuf, Ez, 2, 0, Bz, Ez, prpack);
        gemm(prpack, wb_out, HWz, Ez, Ez, 1,
             out_b, nullptr, nullptr, 0,
             out + (size_t)s * HWz, Sz * HWz, 1, 0, Bz, nullptr, 0, 0);

        // hidden = t[:,1] -> packed for next step's he GEMM
        pack(xbuf, Ez, 2, 1, Bz, Ez, hidpack);
    }
}
